// LocalAttention_87316685128260
// MI455X (gfx1250) — matrix operand order repaired and checked by reading, not compiled, not run on hardware
//
#include <hip/hip_runtime.h>

typedef __attribute__((ext_vector_type(16))) __bf16 v16bf;
typedef __attribute__((ext_vector_type(8)))  __bf16 v8bf;
typedef __attribute__((ext_vector_type(8)))  float  v8f;

static constexpr int BATCH = 4;
static constexpr int SEQ   = 2048;
static constexpr int DIM   = 1024;
static constexpr int BAND  = 384;     // 3 key tiles of 128 cover [q0-128, q0+255]
static constexpr int PA    = 40;      // LDS pitch for A tile (bf16 elems), 80B rows -> 16B aligned
static constexpr int PB    = 136;     // LDS pitch for B tile, 272B rows -> 16B aligned

__device__ __forceinline__ unsigned short f2bf(float f) {
  unsigned int u = __float_as_uint(f);
  unsigned int r = u + 0x7FFFu + ((u >> 16) & 1u);   // round-to-nearest-even
  return (unsigned short)(r >> 16);
}

// LDS byte offset of a generic __shared__ pointer (low 32 bits of the shared aperture address)
__device__ __forceinline__ unsigned lds_off(const void* p) {
  return (unsigned)(uintptr_t)p;
}

// async DMA: 16 bytes global -> LDS per lane (ASYNCcnt-tracked, bypasses VGPRs)
__device__ __forceinline__ void async_b128(unsigned lds, const void* gaddr) {
  asm volatile("global_load_async_to_lds_b128 %0, %1, off"
               :: "v"(lds), "v"(gaddr) : "memory");
}
__device__ __forceinline__ void wait_async() {
  asm volatile("s_wait_asynccnt 0x0" ::: "memory");
}

// ---- WMMA fragment loads from LDS (layouts per CDNA5 ISA 7.12.2) ------------
// A 16x32 bf16: lane L (0..15): row=m0+L, K = {0..7,16..23}; lane L+16: K = {8..15,24..31}
__device__ __forceinline__ v16bf ldA(const __bf16* As, int m0) {
  const int lane = threadIdx.x & 31;
  const __bf16* p = As + (m0 + (lane & 15)) * PA + ((lane >> 4) * 8);
  v8bf x = *(const v8bf*)(p);
  v8bf y = *(const v8bf*)(p + 16);
  return __builtin_shufflevector(x, y, 0,1,2,3,4,5,6,7,8,9,10,11,12,13,14,15);
}
// B 32x16 bf16: lane L (0..15): col=n0+L, K = {0..7,16..23}; lane L+16: K = {8..15,24..31}
__device__ __forceinline__ v16bf ldB(const __bf16* Bs, int n0) {
  const int lane = threadIdx.x & 31;
  const __bf16* p = Bs + ((lane >> 4) * 8) * PB + n0 + (lane & 15);
  v16bf r;
#pragma unroll
  for (int j = 0; j < 8; ++j) { r[j] = p[j * PB]; r[8 + j] = p[(16 + j) * PB]; }
  return r;
}

// one 128x128x32 step: wave (wm,wn) owns 64x32 = 4x2 WMMA tiles
__device__ __forceinline__ void mma_step(const __bf16* As, const __bf16* Bs,
                                         int wm, int wn, v8f (&acc)[4][2]) {
  v16bf af[4], bfg[2];
#pragma unroll
  for (int sm = 0; sm < 4; ++sm) af[sm] = ldA(As, wm * 64 + sm * 16);
#pragma unroll
  for (int sn = 0; sn < 2; ++sn) bfg[sn] = ldB(Bs, wn * 32 + sn * 16);
#pragma unroll
  for (int sm = 0; sm < 4; ++sm)
#pragma unroll
    for (int sn = 0; sn < 2; ++sn)
      acc[sm][sn] = __builtin_amdgcn_wmma_f32_16x16x32_bf16(
          false, af[sm], false, bfg[sn], (short)0, acc[sm][sn], false, false);
}

// ---- LDS staging ------------------------------------------------------------
// A: 128 rows x 32 cols, global row-major (async DMA)
__device__ __forceinline__ void stageA_async(__bf16* As, const __bf16* Ag, int lda) {
  const int t = threadIdx.x;
  const int r = t >> 2, c = (t & 3) * 8;
  async_b128(lds_off(As + r * PA + c),        Ag + (size_t)r * lda + c);
  async_b128(lds_off(As + (r + 64) * PA + c), Ag + (size_t)(r + 64) * lda + c);
}
// B: 32 rows x 128 cols, global row-major (weights) (async DMA)
__device__ __forceinline__ void stageB_async(__bf16* Bs, const __bf16* Bg, int ldb) {
  const int t = threadIdx.x;
  const int r = t >> 4, c = (t & 15) * 8;
  async_b128(lds_off(Bs + r * PB + c),        Bg + (size_t)r * ldb + c);
  async_b128(lds_off(Bs + (r + 16) * PB + c), Bg + (size_t)(r + 16) * ldb + c);
}
// B with clamped global rows (V rows outside [0,SEQ) -> clamp; P is zero there)
__device__ __forceinline__ void stageB_rows_async(__bf16* Bs, const __bf16* Vb, int ldb,
                                                  int row0, int col0) {
  const int t = threadIdx.x;
  const int r = t >> 4, c = (t & 15) * 8;
#pragma unroll
  for (int it = 0; it < 2; ++it) {
    const int rr = r + it * 16;
    int gr = row0 + rr;
    gr = gr < 0 ? 0 : (gr > SEQ - 1 ? SEQ - 1 : gr);
    async_b128(lds_off(Bs + rr * PB + c), Vb + (size_t)gr * ldb + col0 + c);
  }
}
// B transposed: 128 keys x 32 d-cols of K (keys row-major) -> Bs[d][key] (element scatter)
__device__ __forceinline__ void stageBT(__bf16* Bs, const __bf16* Kb, int ldb, int key0) {
  const int t = threadIdx.x;
  const int kk = t >> 2, c = (t & 3) * 8;
#pragma unroll
  for (int it = 0; it < 2; ++it) {
    const int lk = kk + it * 64;
    const int key = key0 + lk;
    v8bf val;
    if (key >= 0 && key < SEQ) {
      val = *(const v8bf*)(Kb + (size_t)key * ldb + c);
    } else {
#pragma unroll
      for (int j = 0; j < 8; ++j) val[j] = (__bf16)0.0f;
    }
#pragma unroll
    for (int j = 0; j < 8; ++j) Bs[(c + j) * PB + lk] = val[j];
  }
}

// ---- kernels ----------------------------------------------------------------
__global__ void cast_bf16_kernel(const float* __restrict__ src,
                                 unsigned short* __restrict__ dst, int n4) {
  const int i = (blockIdx.x * blockDim.x + threadIdx.x) * 4;
  if (i < n4) {
    float4 f = *(const float4*)(src + i);
    ushort4 o;
    o.x = f2bf(f.x); o.y = f2bf(f.y); o.z = f2bf(f.z); o.w = f2bf(f.w);
    *(ushort4*)(dst + i) = o;
  }
}

// double-buffered LDS + async DMA staging; next tile issued after the barrier so its
// writes overlap this tile's WMMAs and cannot race other waves' previous-tile reads
template <bool BF16OUT>
__global__ __launch_bounds__(256) void gemm_bias_kernel(
    const __bf16* __restrict__ A, int lda, const __bf16* __restrict__ Bm, int ldb,
    const float* __restrict__ bias, void* __restrict__ Cout, int ldc, int K) {
  __shared__ __bf16 As[2][128 * PA];
  __shared__ __bf16 Bs[2][32 * PB];
  const int row0 = blockIdx.x * 128, col0 = blockIdx.y * 128;
  const int wid = threadIdx.x >> 5, wm = wid >> 2, wn = wid & 3;
  v8f acc[4][2] = {};
  const int nk = K >> 5;
  stageA_async(As[0], A + (size_t)row0 * lda, lda);
  stageB_async(Bs[0], Bm + col0, ldb);
  for (int i = 0; i < nk; ++i) {
    const int cur = i & 1;
    wait_async();        // my writes of tile i are in LDS
    __syncthreads();     // everyone's writes of tile i in LDS; tile i-1 reads all done
    if (i + 1 < nk) {    // prefetch tile i+1 into the other buffer, overlapping compute
      stageA_async(As[cur ^ 1], A + (size_t)row0 * lda + (i + 1) * 32, lda);
      stageB_async(Bs[cur ^ 1], Bm + (size_t)(i + 1) * 32 * ldb + col0, ldb);
    }
    mma_step(As[cur], Bs[cur], wm, wn, acc);
  }
  const int lane = threadIdx.x & 31, cn = lane & 15, mh = (lane >> 4) * 8;
#pragma unroll
  for (int sm = 0; sm < 4; ++sm)
#pragma unroll
    for (int sn = 0; sn < 2; ++sn)
#pragma unroll
      for (int r = 0; r < 8; ++r) {
        const int row = row0 + wm * 64 + sm * 16 + mh + r;
        const int col = col0 + wn * 32 + sn * 16 + cn;
        const float v = acc[sm][sn][r] + bias[col];
        if (BF16OUT) ((unsigned short*)Cout)[(size_t)row * ldc + col] = f2bf(v);
        else         ((float*)Cout)[(size_t)row * ldc + col] = v;
      }
}

__global__ __launch_bounds__(256) void scores_kernel(const __bf16* __restrict__ QKV,
                                                     float* __restrict__ S,
                                                     const int* __restrict__ winp) {
  __shared__ __bf16 As[128 * PA];
  __shared__ __bf16 Bs[32 * PB];
  const int qt = blockIdx.x, kt = blockIdx.y, b = blockIdx.z;
  const int W = winp[0];
  const int key0 = qt * 128 - 128 + kt * 128;
  const __bf16* Q  = QKV + ((size_t)b * SEQ + qt * 128) * (3 * DIM);        // Q view
  const __bf16* Kb = QKV + (size_t)b * SEQ * (3 * DIM) + DIM;               // K view
  const int wid = threadIdx.x >> 5, wm = wid >> 2, wn = wid & 3;
  v8f acc[4][2] = {};
  for (int kb = 0; kb < DIM; kb += 32) {
    stageA_async(As, Q + kb, 3 * DIM);
    stageBT(Bs, Kb + kb, 3 * DIM, key0);
    wait_async();
    __syncthreads();
    mma_step(As, Bs, wm, wn, acc);
    __syncthreads();
  }
  const float scale = 0.03125f;  // D^-0.5
  const int lane = threadIdx.x & 31, cn = lane & 15, mh = (lane >> 4) * 8;
#pragma unroll
  for (int sm = 0; sm < 4; ++sm)
#pragma unroll
    for (int sn = 0; sn < 2; ++sn)
#pragma unroll
      for (int r = 0; r < 8; ++r) {
        const int qrow  = qt * 128 + wm * 64 + sm * 16 + mh + r;
        const int kslot = kt * 128 + wn * 32 + sn * 16 + cn;
        const int key = qt * 128 - 128 + kslot;
        int d = qrow - key; d = d < 0 ? -d : d;
        const bool valid = (key >= 0) && (key < SEQ) && (d <= W);
        S[((size_t)b * SEQ + qrow) * BAND + kslot] = valid ? acc[sm][sn][r] * scale : -1e30f;
      }
}

__global__ __launch_bounds__(256) void softmax_kernel(const float* __restrict__ S,
                                                      unsigned short* __restrict__ P) {
  const int wid = threadIdx.x >> 5, lane = threadIdx.x & 31;
  const int row = blockIdx.x * 8 + wid;  // one wave per band row
  const float* sr = S + (size_t)row * BAND;
  float v[12];
  float m = -3.0e38f;
#pragma unroll
  for (int j = 0; j < 12; ++j) { v[j] = sr[lane + j * 32]; m = fmaxf(m, v[j]); }
#pragma unroll
  for (int o = 16; o > 0; o >>= 1) m = fmaxf(m, __shfl_xor(m, o, 32));
  float s = 0.0f;
#pragma unroll
  for (int j = 0; j < 12; ++j) { v[j] = __expf(v[j] - m); s += v[j]; }
#pragma unroll
  for (int o = 16; o > 0; o >>= 1) s += __shfl_xor(s, o, 32);
  const float inv = 1.0f / s;
#pragma unroll
  for (int j = 0; j < 12; ++j) P[(size_t)row * BAND + lane + j * 32] = f2bf(v[j] * inv);
}

__global__ __launch_bounds__(256) void pv_kernel(const __bf16* __restrict__ QKV,
                                                 const __bf16* __restrict__ P,
                                                 unsigned short* __restrict__ AO) {
  __shared__ __bf16 As[128 * PA];
  __shared__ __bf16 Bs[32 * PB];
  const int qt = blockIdx.x, nt = blockIdx.y, b = blockIdx.z;
  const int key0 = qt * 128 - 128;
  const __bf16* Ap = P + ((size_t)b * SEQ + qt * 128) * BAND;
  const __bf16* Vb = QKV + (size_t)b * SEQ * (3 * DIM) + 2 * DIM;  // V view
  const int wid = threadIdx.x >> 5, wm = wid >> 2, wn = wid & 3;
  v8f acc[4][2] = {};
  for (int kb = 0; kb < BAND; kb += 32) {
    stageA_async(As, Ap + kb, BAND);
    stageB_rows_async(Bs, Vb, 3 * DIM, key0 + kb, nt * 128);
    wait_async();
    __syncthreads();
    mma_step(As, Bs, wm, wn, acc);
    __syncthreads();
  }
  const int lane = threadIdx.x & 31, cn = lane & 15, mh = (lane >> 4) * 8;
#pragma unroll
  for (int sm = 0; sm < 4; ++sm)
#pragma unroll
    for (int sn = 0; sn < 2; ++sn)
#pragma unroll
      for (int r = 0; r < 8; ++r) {
        const int row = b * SEQ + qt * 128 + wm * 64 + sm * 16 + mh + r;
        const int col = nt * 128 + wn * 32 + sn * 16 + cn;
        AO[(size_t)row * DIM + col] = f2bf(acc[sm][sn][r]);
      }
}

// ---- launch -----------------------------------------------------------------
extern "C" void kernel_launch(void* const* d_in, const int* in_sizes, int n_in,
                              void* d_out, int out_size, void* d_ws, size_t ws_size,
                              hipStream_t stream) {
  (void)in_sizes; (void)n_in; (void)out_size; (void)ws_size;
  const float* x      = (const float*)d_in[0];
  const float* w_qkv  = (const float*)d_in[1];
  const float* b_qkv  = (const float*)d_in[2];
  const float* w_proj = (const float*)d_in[3];
  const float* b_proj = (const float*)d_in[4];
  const int*   winp   = (const int*)d_in[5];

  char* ws = (char*)d_ws;
  size_t off = 0;
  auto take = [&](size_t bytes) -> char* {
    char* p = ws + off;
    off = (off + bytes + 255) & ~(size_t)255;
    return p;
  };
  const size_t ROWS = (size_t)BATCH * SEQ;                 // 8192
  __bf16* xh      = (__bf16*)take(ROWS * DIM * 2);
  __bf16* wqkvh   = (__bf16*)take((size_t)DIM * 3 * DIM * 2);
  __bf16* wprojh  = (__bf16*)take((size_t)DIM * DIM * 2);
  __bf16* QKV     = (__bf16*)take(ROWS * 3 * DIM * 2);
  float*  S       = (float*)take(ROWS * BAND * 4);
  unsigned short* P  = (unsigned short*)take(ROWS * BAND * 2);
  unsigned short* AO = (unsigned short*)take(ROWS * DIM * 2);

  // 1) fp32 -> bf16 casts
  cast_bf16_kernel<<<(int)(ROWS * DIM / 4 / 256), 256, 0, stream>>>(x, (unsigned short*)xh, (int)(ROWS * DIM));
  cast_bf16_kernel<<<DIM * 3 * DIM / 4 / 256, 256, 0, stream>>>(w_qkv, (unsigned short*)wqkvh, DIM * 3 * DIM);
  cast_bf16_kernel<<<DIM * DIM / 4 / 256, 256, 0, stream>>>(w_proj, (unsigned short*)wprojh, DIM * DIM);

  // 2) QKV projection (bf16 WMMA, fp32 accum) + bias
  gemm_bias_kernel<true><<<dim3((int)(ROWS / 128), 3 * DIM / 128), 256, 0, stream>>>(
      xh, DIM, wqkvh, 3 * DIM, b_qkv, QKV, 3 * DIM, DIM);

  // 3) banded scores: per (q-tile, k-tile, batch)
  scores_kernel<<<dim3(SEQ / 128, 3, BATCH), 256, 0, stream>>>(QKV, S, winp);

  // 4) band softmax, one wave per row
  softmax_kernel<<<(int)(ROWS / 8), 256, 0, stream>>>(S, P);

  // 5) P @ V over the band
  pv_kernel<<<dim3(SEQ / 128, DIM / 128, BATCH), 256, 0, stream>>>(QKV, (const __bf16*)P, AO);

  // 6) output projection + bias -> fp32
  gemm_bias_kernel<false><<<dim3((int)(ROWS / 128), DIM / 128), 256, 0, stream>>>(
      (const __bf16*)AO, DIM, wprojh, DIM, b_proj, d_out, DIM, DIM);
}
